// BDH_Life_57449482551799
// MI455X (gfx1250) — matrix operand order repaired and checked by reading, not compiled, not run on hardware
//
#include <hip/hip_runtime.h>
#include <hip/hip_bf16.h>

// ---------------- problem constants ----------------
constexpr int B  = 8;
constexpr int T  = 1024;
constexpr int D  = 256;
constexpr int NH = 4;
constexpr int N  = 256;          // sparse head dim
constexpr int NHN = NH * N;      // 1024
constexpr int NLAYER = 3;
constexpr float EPS = 1e-5f;

typedef __attribute__((ext_vector_type(16))) __bf16 v16bf;
typedef __attribute__((ext_vector_type(8)))  float  v8f;

// ---------------- WMMA fragment helpers (wave32, 16x16x32 bf16) ----------------
// A-matrix 16x32 (MxK): lane m=lane&15, half h=lane>>4.
// element e<8 : k = 8h + e ; element e>=8 : k = 16 + 8h + (e-8)   (ISA 7.12.2)
static __device__ __forceinline__ v16bf load_afrag(const __bf16* base, int ld, int lane) {
  const int m = lane & 15, h = (lane >> 4) & 1;
  const __bf16* r = base + (size_t)m * ld;
  v16bf a;
#pragma unroll
  for (int i = 0; i < 8; ++i) a[i] = r[8 * h + i];
#pragma unroll
  for (int i = 0; i < 8; ++i) a[8 + i] = r[16 + 8 * h + i];
  return a;
}

// B-matrix 32x16 (KxN): lane n=lane&15, half h: element e<8 : k = 8h + e ; e>=8 : k = 16 + 8h + (e-8).
// Memory holds B transposed (N rows x K cols, row-major) => 8-element runs per lane.
static __device__ __forceinline__ v16bf load_bfrag_t(const __bf16* baseT, int ld, int lane) {
  const int n = lane & 15, h = (lane >> 4) & 1;
  const __bf16* r = baseT + (size_t)n * ld;
  v16bf b;
#pragma unroll
  for (int i = 0; i < 8; ++i) b[i] = r[8 * h + i];
#pragma unroll
  for (int i = 0; i < 8; ++i) b[8 + i] = r[16 + 8 * h + i];
  return b;
}

static __device__ __forceinline__ v8f wmma_bf16(v16bf a, v16bf b, v8f c) {
  return __builtin_amdgcn_wmma_f32_16x16x32_bf16(false, a, false, b, (short)0, c, false, false);
}

// ---------------- prep kernels ----------------
__global__ void freqs_kernel(float* __restrict__ cosT, float* __restrict__ sinT) {
  int i = blockIdx.x * blockDim.x + threadIdx.x;     // T*128
  if (i >= T * 128) return;
  int t = i >> 7, j = i & 127;
  float pos = (j < 64) ? (float)(t & 31) : (float)(t >> 5);
  int jj = j & 63;
  float freq = __powf(10000.f, -(float)jj / 64.f);
  float ang = pos * freq;
  cosT[i] = __cosf(ang);
  sinT[i] = __sinf(ang);
}

// src: nmat x R x C fp32 row-major -> dst: nmat x C x R bf16 (transposed)
__global__ void transp_bf16_kernel(const float* __restrict__ src, __bf16* __restrict__ dst,
                                   int R, int C, int nmat) {
  size_t i = (size_t)blockIdx.x * blockDim.x + threadIdx.x;
  size_t tot = (size_t)nmat * R * C;
  if (i >= tot) return;
  int rc = (int)(i % (size_t)(R * C));
  int m  = (int)(i / (size_t)(R * C));
  int r = rc / C, c = rc % C;
  dst[(size_t)m * R * C + (size_t)c * R + r] = (__bf16)src[i];
}

// x = LN(inputs[...,None] @ in_w.T + in_b) ; one wave per (b,t) row
// Also writes bf16 copy (B,T,D) and transposed bf16 copy (B,D,T) for attention V.
__global__ __launch_bounds__(256) void inproj_kernel(const float* __restrict__ inp,
                                                     const float* __restrict__ inw,
                                                     const float* __restrict__ inb,
                                                     float* __restrict__ x,
                                                     __bf16* __restrict__ xbf,
                                                     __bf16* __restrict__ xbfT) {
  int row = blockIdx.x * 8 + (threadIdx.x >> 5);   // flat b*T + t
  int lane = threadIdx.x & 31;
  int b = row >> 10, t = row & (T - 1);
  float iv = inp[row];
  float vals[8];
  float s1 = 0.f, s2 = 0.f;
#pragma unroll
  for (int i = 0; i < 8; ++i) {
    int d = lane + 32 * i;
    float v = iv * inw[d] + inb[d];
    vals[i] = v; s1 += v; s2 += v * v;
  }
#pragma unroll
  for (int o = 1; o < 32; o <<= 1) { s1 += __shfl_xor(s1, o, 32); s2 += __shfl_xor(s2, o, 32); }
  float mu = s1 * (1.f / D);
  float rstd = rsqrtf(s2 * (1.f / D) - mu * mu + EPS);
#pragma unroll
  for (int i = 0; i < 8; ++i) {
    int d = lane + 32 * i;
    float v = (vals[i] - mu) * rstd;
    x[(size_t)row * D + d] = v;
    xbf[(size_t)row * D + d] = (__bf16)v;
    xbfT[((size_t)b * D + d) * T + t] = (__bf16)v;
  }
}

// ---------------- encoder GEMM + ReLU + RoPE ----------------
// grid (B*NH, T/32), 64 threads (2 waves); wave computes 16 rows x N cols.
__global__ __launch_bounds__(64) void enc_kernel(const __bf16* __restrict__ xbf,
                                                 const __bf16* __restrict__ encT,  // NH x (N x D)
                                                 float* __restrict__ xsp,
                                                 __bf16* __restrict__ qr,
                                                 const float* __restrict__ cosT,
                                                 const float* __restrict__ sinT) {
  const int lane = threadIdx.x & 31, wave = threadIdx.x >> 5;
  const int bh = blockIdx.x, b = bh / NH, h = bh % NH;
  const int t0 = blockIdx.y * 32 + wave * 16;
  const __bf16* A = xbf + ((size_t)b * T + t0) * D;
  const __bf16* Bm = encT + (size_t)h * N * D;

  v8f zero = {};
  v8f acc[16];
#pragma unroll
  for (int f = 0; f < 16; ++f) acc[f] = zero;

  for (int k = 0; k < 8; ++k) {
    v16bf a = load_afrag(A + 32 * k, D, lane);
#pragma unroll
    for (int f = 0; f < 16; ++f) {
      v16bf bb = load_bfrag_t(Bm + (size_t)(16 * f) * D + 32 * k, D, lane);
      acc[f] = wmma_bf16(a, bb, acc[f]);
    }
  }

  const int n0 = lane & 15, hh = (lane >> 4) & 1;
#pragma unroll
  for (int f = 0; f < 16; ++f) {
#pragma unroll
    for (int v = 0; v < 8; ++v) {
      float val = fmaxf(acc[f][v], 0.f);          // relu -> x_sparse
      int t = t0 + v + 8 * hh;
      int n = 16 * f + n0;
      size_t idx = ((size_t)bh * T + t) * N + n;
      xsp[idx] = val;
      // RoPE: complex pair (n even, n odd) lives in adjacent lanes
      float other = __shfl_xor(val, 1, 32);
      int j = n >> 1;
      float c = cosT[t * 128 + j], s = sinT[t * 128 + j];
      float ro = (n & 1) ? (other * s + val * c)   // imag lane
                         : (val * c - other * s);  // real lane
      qr[idx] = (__bf16)ro;
    }
  }
}

// ---------------- flash attention + LN(yKV) ----------------
// grid (B*NH, T/64), 128 threads (4 waves); each wave owns a 16-row Q tile.
// K chunk (32 x N) and V chunk (transposed, D x 32) staged in LDS per block.
__global__ __launch_bounds__(128) void attn_kernel(const __bf16* __restrict__ QR,
                                                   const __bf16* __restrict__ xT,   // (B, D, T) bf16
                                                   __bf16* __restrict__ ykv) {
  const int tid = threadIdx.x;
  const int lane = tid & 31, wave = tid >> 5;
  const int bh = blockIdx.x, b = bh / NH;
  const int q0 = blockIdx.y * 64 + wave * 16;
  const __bf16* Q  = QR + (size_t)bh * T * N;
  const __bf16* Vt = xT + (size_t)b * D * T;      // row d, col t

  __shared__ __bf16 ks[32 * N];        // key chunk, row s, col n      (16 KB)
  __shared__ __bf16 vs[D * 32];        // V chunk transposed, row d, col s (16 KB)
  __shared__ __bf16 pbuf[4][16 * 32];  // per-wave P tile               (4 KB)

  v16bf qa[8];
#pragma unroll
  for (int k = 0; k < 8; ++k) qa[k] = load_afrag(Q + (size_t)q0 * N + 32 * k, N, lane);

  v8f zero = {};
  v8f acc[16];
#pragma unroll
  for (int f = 0; f < 16; ++f) acc[f] = zero;
  float mrow[8], lrow[8];
#pragma unroll
  for (int v = 0; v < 8; ++v) { mrow[v] = -3.0e38f; lrow[v] = 0.f; }

  const float scale = 0.0625f;             // 1/sqrt(N)
  const int n0 = lane & 15, hh = (lane >> 4) & 1;

  for (int s0 = 0; s0 < T; s0 += 32) {
    // ---- cooperative staging (vectorized 16B copies) ----
    {
      // K chunk: 32 rows x 256 cols; thread -> row s = tid>>2, quarter q = tid&3 (64 elems)
      int s = tid >> 2, q = tid & 3;
      const uint4* src = (const uint4*)(Q + (size_t)(s0 + s) * N + q * 64);
      uint4* dst = (uint4*)(&ks[s * N + q * 64]);
#pragma unroll
      for (int i = 0; i < 8; ++i) dst[i] = src[i];
      // V chunk: 256 d-rows x 32 s-cols from xT; thread -> rows 2*tid, 2*tid+1
#pragma unroll
      for (int r = 0; r < 2; ++r) {
        int d = tid * 2 + r;
        const uint4* vsrc = (const uint4*)(Vt + (size_t)d * T + s0);
        uint4* vdst = (uint4*)(&vs[d * 32]);
#pragma unroll
        for (int i = 0; i < 4; ++i) vdst[i] = vsrc[i];
      }
    }
    if (s0 + 32 < T) __builtin_prefetch(Q + (size_t)(s0 + 32) * N, 0, 1);
    __syncthreads();

    // ---- two 16x16 score tiles (K == Q after RoPE) ----
    v8f sc[2];
#pragma unroll
    for (int st = 0; st < 2; ++st) {
      v8f sa = zero;
#pragma unroll
      for (int k = 0; k < 8; ++k) {
        v16bf bb = load_bfrag_t(&ks[(16 * st) * N + 32 * k], N, lane);
        sa = wmma_bf16(qa[k], bb, sa);
      }
      sc[st] = sa * scale;
    }
    // ---- online softmax (row r = v + 8*hh, reduced across 16 lanes of the half) ----
    float corr[8];
#pragma unroll
    for (int v = 0; v < 8; ++v) {
      float mx = fmaxf(sc[0][v], sc[1][v]);
#pragma unroll
      for (int o = 1; o < 16; o <<= 1) mx = fmaxf(mx, __shfl_xor(mx, o, 16));
      float mnew = fmaxf(mrow[v], mx);
      corr[v] = __expf(mrow[v] - mnew);
      mrow[v] = mnew;
      float p0 = __expf(sc[0][v] - mnew);
      float p1 = __expf(sc[1][v] - mnew);
      sc[0][v] = p0; sc[1][v] = p1;
      float rs = p0 + p1;
#pragma unroll
      for (int o = 1; o < 16; o <<= 1) rs += __shfl_xor(rs, o, 16);
      lrow[v] = lrow[v] * corr[v] + rs;
    }
#pragma unroll
    for (int f = 0; f < 16; ++f)
#pragma unroll
      for (int v = 0; v < 8; ++v) acc[f][v] *= corr[v];

    // ---- re-layout P (C-frag) -> A-frag through per-wave LDS ----
    __bf16* pb = &pbuf[wave][0];
#pragma unroll
    for (int st = 0; st < 2; ++st)
#pragma unroll
      for (int v = 0; v < 8; ++v)
        pb[(v + 8 * hh) * 32 + 16 * st + n0] = (__bf16)sc[st][v];
    v16bf pa = load_afrag(pb, 32, lane);

    // ---- acc += P @ V  (V chunk transposed in LDS: row d, col s) ----
#pragma unroll
    for (int nt = 0; nt < 16; ++nt) {
      v16bf vb = load_bfrag_t(&vs[(16 * nt) * 32], 32, lane);
      acc[nt] = wmma_bf16(pa, vb, acc[nt]);
    }
    __syncthreads();    // before next chunk overwrites ks/vs
  }

  // normalize by l, then LayerNorm over D
  float mu[8], rstd[8];
#pragma unroll
  for (int v = 0; v < 8; ++v) {
    float inv = 1.f / lrow[v];
    float s1 = 0.f, s2 = 0.f;
#pragma unroll
    for (int f = 0; f < 16; ++f) {
      float val = acc[f][v] * inv;
      acc[f][v] = val;
      s1 += val; s2 += val * val;
    }
#pragma unroll
    for (int o = 1; o < 16; o <<= 1) { s1 += __shfl_xor(s1, o, 16); s2 += __shfl_xor(s2, o, 16); }
    float m = s1 * (1.f / D);
    mu[v] = m;
    rstd[v] = rsqrtf(s2 * (1.f / D) - m * m + EPS);
  }
#pragma unroll
  for (int f = 0; f < 16; ++f)
#pragma unroll
    for (int v = 0; v < 8; ++v) {
      int t = q0 + v + 8 * hh;
      ykv[((size_t)bh * T + t) * D + 16 * f + n0] = (__bf16)((acc[f][v] - mu[v]) * rstd[v]);
    }
}

// ---------------- encoder_v GEMM + ReLU + multiply by x_sparse ----------------
__global__ __launch_bounds__(64) void encv_kernel(const __bf16* __restrict__ ykv,
                                                  const __bf16* __restrict__ encvT, // NH x (N x D)
                                                  const float* __restrict__ xsp,
                                                  __bf16* __restrict__ xy) {
  const int lane = threadIdx.x & 31, wave = threadIdx.x >> 5;
  const int bh = blockIdx.x, b = bh / NH, h = bh % NH;
  const int t0 = blockIdx.y * 32 + wave * 16;
  const __bf16* A = ykv + ((size_t)bh * T + t0) * D;
  const __bf16* Bm = encvT + (size_t)h * N * D;

  v8f zero = {};
  v8f acc[16];
#pragma unroll
  for (int f = 0; f < 16; ++f) acc[f] = zero;

  for (int k = 0; k < 8; ++k) {
    v16bf a = load_afrag(A + 32 * k, D, lane);
#pragma unroll
    for (int f = 0; f < 16; ++f) {
      v16bf bb = load_bfrag_t(Bm + (size_t)(16 * f) * D + 32 * k, D, lane);
      acc[f] = wmma_bf16(a, bb, acc[f]);
    }
  }

  const int n0 = lane & 15, hh = (lane >> 4) & 1;
#pragma unroll
  for (int f = 0; f < 16; ++f)
#pragma unroll
    for (int v = 0; v < 8; ++v) {
      int t = t0 + v + 8 * hh;
      int n = 16 * f + n0;
      float y = fmaxf(acc[f][v], 0.f);
      float val = y * xsp[((size_t)bh * T + t) * N + n];
      xy[((size_t)b * T + t) * NHN + (size_t)h * N + n] = (__bf16)val;
    }
}

// ---------------- decoder GEMM + LN + residual + LN (x updated in place) ----------------
// grid B*T/16 blocks, 1 wave each; 16 rows x D cols, K = NH*N = 1024.
__global__ __launch_bounds__(32) void dec_kernel(const __bf16* __restrict__ xy,
                                                 const __bf16* __restrict__ decT,  // D x NHN
                                                 float* __restrict__ x,
                                                 __bf16* __restrict__ xbf,
                                                 __bf16* __restrict__ xbfT) {
  const int lane = threadIdx.x & 31;
  const int r0 = blockIdx.x * 16;            // flat (b*T + t) row
  const __bf16* A = xy + (size_t)r0 * NHN;

  v8f zero = {};
  v8f acc[16];
#pragma unroll
  for (int f = 0; f < 16; ++f) acc[f] = zero;

  for (int k = 0; k < 32; ++k) {
    v16bf a = load_afrag(A + 32 * k, NHN, lane);
#pragma unroll
    for (int f = 0; f < 16; ++f) {
      v16bf bb = load_bfrag_t(decT + (size_t)(16 * f) * NHN + 32 * k, NHN, lane);
      acc[f] = wmma_bf16(a, bb, acc[f]);
    }
  }

  const int n0 = lane & 15, hh = (lane >> 4) & 1;
  // LN(yMLP)
  float mu1[8], r1[8];
#pragma unroll
  for (int v = 0; v < 8; ++v) {
    float s1 = 0.f, s2 = 0.f;
#pragma unroll
    for (int f = 0; f < 16; ++f) { float z = acc[f][v]; s1 += z; s2 += z * z; }
#pragma unroll
    for (int o = 1; o < 16; o <<= 1) { s1 += __shfl_xor(s1, o, 16); s2 += __shfl_xor(s2, o, 16); }
    float m = s1 * (1.f / D);
    mu1[v] = m;
    r1[v] = rsqrtf(s2 * (1.f / D) - m * m + EPS);
  }
  // z = x + LN(yMLP); then x = LN(z)
#pragma unroll
  for (int v = 0; v < 8; ++v) {
    int row = r0 + v + 8 * hh;
    int bb_ = row >> 10, tt_ = row & (T - 1);
    float s1 = 0.f, s2 = 0.f;
#pragma unroll
    for (int f = 0; f < 16; ++f) {
      float z = (acc[f][v] - mu1[v]) * r1[v] + x[(size_t)row * D + 16 * f + n0];
      acc[f][v] = z; s1 += z; s2 += z * z;
    }
#pragma unroll
    for (int o = 1; o < 16; o <<= 1) { s1 += __shfl_xor(s1, o, 16); s2 += __shfl_xor(s2, o, 16); }
    float m = s1 * (1.f / D);
    float rs = rsqrtf(s2 * (1.f / D) - m * m + EPS);
#pragma unroll
    for (int f = 0; f < 16; ++f) {
      float xn = (acc[f][v] - m) * rs;
      int col = 16 * f + n0;
      x[(size_t)row * D + col] = xn;
      xbf[(size_t)row * D + col] = (__bf16)xn;
      xbfT[((size_t)bb_ * D + col) * T + tt_] = (__bf16)xn;
    }
  }
}

// ---------------- output projection ----------------
__global__ __launch_bounds__(256) void out_kernel(const float* __restrict__ x,
                                                  const float* __restrict__ ow,
                                                  const float* __restrict__ ob,
                                                  float* __restrict__ out) {
  int row = blockIdx.x * 8 + (threadIdx.x >> 5);
  int lane = threadIdx.x & 31;
  float s = 0.f;
#pragma unroll
  for (int i = 0; i < 8; ++i) s += x[(size_t)row * D + lane + 32 * i] * ow[lane + 32 * i];
#pragma unroll
  for (int o = 1; o < 32; o <<= 1) s += __shfl_xor(s, o, 32);
  if (lane == 0) out[row] = s + ob[0];
}

// ---------------- host orchestration ----------------
extern "C" void kernel_launch(void* const* d_in, const int* in_sizes, int n_in,
                              void* d_out, int out_size, void* d_ws, size_t ws_size,
                              hipStream_t stream) {
  (void)in_sizes; (void)n_in; (void)out_size; (void)ws_size;
  const float* inp  = (const float*)d_in[0];
  const float* inw  = (const float*)d_in[1];
  const float* inb  = (const float*)d_in[2];
  const float* enc  = (const float*)d_in[3];
  const float* encv = (const float*)d_in[4];
  const float* dec  = (const float*)d_in[5];
  const float* ow   = (const float*)d_in[6];
  const float* ob   = (const float*)d_in[7];
  float* out = (float*)d_out;

  char* w = (char*)d_ws;
  auto alloc = [&](size_t bytes) -> void* {
    void* p = (void*)w;
    w += (bytes + 255) & ~(size_t)255;
    return p;
  };
  float*  cosT  = (float*)alloc((size_t)T * 128 * sizeof(float));
  float*  sinT  = (float*)alloc((size_t)T * 128 * sizeof(float));
  __bf16* encT  = (__bf16*)alloc((size_t)NH * N * D * sizeof(__bf16));
  __bf16* encvT = (__bf16*)alloc((size_t)NH * N * D * sizeof(__bf16));
  __bf16* decT  = (__bf16*)alloc((size_t)D * NHN * sizeof(__bf16));
  float*  x     = (float*)alloc((size_t)B * T * D * sizeof(float));
  __bf16* xbf   = (__bf16*)alloc((size_t)B * T * D * sizeof(__bf16));
  __bf16* xbfT  = (__bf16*)alloc((size_t)B * D * T * sizeof(__bf16));
  float*  xsp   = (float*)alloc((size_t)B * NH * T * N * sizeof(float));
  __bf16* qr    = (__bf16*)alloc((size_t)B * NH * T * N * sizeof(__bf16));
  __bf16* ykv   = (__bf16*)alloc((size_t)B * NH * T * D * sizeof(__bf16));
  __bf16* xy    = (__bf16*)alloc((size_t)B * T * NHN * sizeof(__bf16));

  freqs_kernel<<<(T * 128 + 255) / 256, 256, 0, stream>>>(cosT, sinT);
  transp_bf16_kernel<<<(NH * D * N + 255) / 256, 256, 0, stream>>>(enc,  encT,  D,   N, NH);
  transp_bf16_kernel<<<(NH * D * N + 255) / 256, 256, 0, stream>>>(encv, encvT, D,   N, NH);
  transp_bf16_kernel<<<(NHN * D + 255) / 256, 256, 0, stream>>>(dec,  decT,  NHN, D, 1);
  inproj_kernel<<<(B * T) / 8, 256, 0, stream>>>(inp, inw, inb, x, xbf, xbfT);

  for (int layer = 0; layer < NLAYER; ++layer) {
    enc_kernel<<<dim3(B * NH, T / 32), 64, 0, stream>>>(xbf, encT, xsp, qr, cosT, sinT);
    attn_kernel<<<dim3(B * NH, T / 64), 128, 0, stream>>>(qr, xbfT, ykv);
    encv_kernel<<<dim3(B * NH, T / 32), 64, 0, stream>>>(ykv, encvT, xsp, xy);
    dec_kernel<<<(B * T) / 16, 32, 0, stream>>>(xy, decT, x, xbf, xbfT);
  }
  out_kernel<<<(B * T) / 8, 256, 0, stream>>>(x, ow, ob, out);
}
